// MTREncoder_52493090292057
// MI455X (gfx1250) — matrix operand order repaired and checked by reading, not compiled, not run on hardware
//
#include <hip/hip_runtime.h>
#include <math.h>

// ---------------- problem constants ----------------
#define BB 16
#define NOBJ 64
#define TT 21
#define NMAP 768
#define PPTS 20
#define CAA 29
#define CMM 9
#define DIM 256
#define NHEAD 8
#define NLAYER 6
#define KNB 16
#define NTOK (NOBJ + NMAP)        // 832
#define HDIM (DIM / NHEAD)        // 32
#define ROWS (BB * NTOK)          // 13312

typedef __attribute__((ext_vector_type(16))) _Float16 v16h;
typedef __attribute__((ext_vector_type(8)))  _Float16 v8h;
typedef __attribute__((ext_vector_type(8)))  float    v8f;
typedef __attribute__((ext_vector_type(4)))  float    f4;

#define LDH 40   // halves per LDS row (32 + 8 pad) = 80 B, 16B-aligned stride

// fast-path staging: two aligned float4 loads per operand (global_load_b128)
__device__ __forceinline__ void stage_fast(const float* __restrict__ rowA,
                                           const float* __restrict__ rowW,
                                           int k0, int skb, float mA, float mB,
                                           float* fa, float* fw)
{
    const f4* pa = (const f4*)(rowA + k0 + skb);
    const f4* pw = (const f4*)(rowW + k0 + skb);
    f4 a0 = pa[0], a1 = pa[1];
    f4 w0 = pw[0], w1 = pw[1];
    fa[0] = a0.x * mA; fa[1] = a0.y * mA; fa[2] = a0.z * mA; fa[3] = a0.w * mA;
    fa[4] = a1.x * mA; fa[5] = a1.y * mA; fa[6] = a1.z * mA; fa[7] = a1.w * mA;
    fw[0] = w0.x * mB; fw[1] = w0.y * mB; fw[2] = w0.z * mB; fw[3] = w0.w * mB;
    fw[4] = w1.x * mB; fw[5] = w1.y * mB; fw[6] = w1.z * mB; fw[7] = w1.w * mB;
}

// fragment load (ISA 7.12.2 layouts) + 2 WMMAs, A fragment reused
__device__ __forceinline__ void wmma_step(const _Float16 (*As)[LDH],
                                          const _Float16 (*Bs)[LDH],
                                          int sm, int sn, int lane,
                                          v8f& acc0, v8f& acc1)
{
    const int arow = sm + (lane & 15);
    const int kh   = (lane >> 4) * 8;   // lanes 0-15: K0..7/16..23 ; 16-31: K8..15/24..31
    v8h a0 = *(const v8h*)&As[arow][kh];
    v8h a1 = *(const v8h*)&As[arow][16 + kh];
    v16h afrag = __builtin_shufflevector(a0, a1,
                    0,1,2,3,4,5,6,7,8,9,10,11,12,13,14,15);

    const int kb2   = (lane >> 4) * 8;  // lanes 0-15: K0..7/16..23 ; 16-31: K8..15/24..31
    const int bcol0 = sn + (lane & 15);
    v8h b00 = *(const v8h*)&Bs[bcol0][kb2];
    v8h b01 = *(const v8h*)&Bs[bcol0][kb2 + 16];
    v16h bf0 = __builtin_shufflevector(b00, b01,
                    0,1,2,3,4,5,6,7,8,9,10,11,12,13,14,15);
    v8h b10 = *(const v8h*)&Bs[bcol0 + 16][kb2];
    v8h b11 = *(const v8h*)&Bs[bcol0 + 16][kb2 + 16];
    v16h bf1 = __builtin_shufflevector(b10, b11,
                    0,1,2,3,4,5,6,7,8,9,10,11,12,13,14,15);

    acc0 = __builtin_amdgcn_wmma_f32_16x16x32_f16(
               false, afrag, false, bf0, (short)0, acc0, false, false);
    acc1 = __builtin_amdgcn_wmma_f32_16x16x32_f16(
               false, afrag, false, bf1, (short)0, acc1, false, false);
}

// =====================================================================
// WMMA GEMM:  C[M,N] = act( (A[M,K] @ W[N,K]^T + bias[N]) * alpha )
// 256 threads = 8 waves; tile 64x64; BK=32; wave = 1 A-frag x 2 B-frags.
// Main loop covers only full 32-wide K tiles with a 1-deep software
// pipeline (next tile's global_load_b128s issue before current WMMAs);
// partial / unaligned K tiles are handled in a separate cold block.
// =====================================================================
__global__ __launch_bounds__(256) void gemm_wmma_kernel(
    const float* __restrict__ A, const float* __restrict__ W,
    const float* __restrict__ bias, float* __restrict__ C,
    int M, int N, int K, float alpha, int do_relu)
{
    __shared__ alignas(16) _Float16 As[64][LDH];   // [m][k]
    __shared__ alignas(16) _Float16 Bs[64][LDH];   // [n][k]

    const int tid  = threadIdx.x;
    const int wave = tid >> 5;
    const int lane = tid & 31;
    const int m0 = blockIdx.y * 64;
    const int n0 = blockIdx.x * 64;
    const int sm = (wave & 3) * 16;
    const int sn = (wave >> 2) * 32;

    // staging: thread stages 8 consecutive k of one A row and one W row
    const int srow = tid >> 2;            // 0..63
    const int skb  = (tid & 3) * 8;       // 0,8,16,24

    const int   gmA  = m0 + srow;
    const int   gnB  = n0 + srow;
    const int   gmAc = gmA < M ? gmA : (M - 1);
    const int   gnBc = gnB < N ? gnB : (N - 1);
    const float mA   = gmA < M ? 1.0f : 0.0f;
    const float mB   = gnB < N ? 1.0f : 0.0f;
    const float* rowA = A + (size_t)gmAc * K;
    const float* rowW = W + (size_t)gnBc * K;

    // full-vector tiles exist only when K%8==0 (then K%32==0 tiles too)
    const int kfull = ((K & 7) == 0) ? (K & ~31) : 0;

    v8f acc0 = {}, acc1 = {};
    float fa[8], fw[8];

    if (kfull > 0) stage_fast(rowA, rowW, 0, skb, mA, mB, fa, fw);

    for (int k0 = 0; k0 < kfull; k0 += 32) {
        v8h ha, hb;
        #pragma unroll
        for (int i = 0; i < 8; ++i) { ha[i] = (_Float16)fa[i]; hb[i] = (_Float16)fw[i]; }
        *(v8h*)&As[srow][skb] = ha;       // ds_store_b128
        *(v8h*)&Bs[srow][skb] = hb;
        __syncthreads();

        // issue next tile's global loads before this tile's WMMAs
        const bool more = (k0 + 32) < kfull;
        float na[8], nw[8];
        if (more) stage_fast(rowA, rowW, k0 + 32, skb, mA, mB, na, nw);

        wmma_step(As, Bs, sm, sn, lane, acc0, acc1);
        __syncthreads();

        if (more) {
            #pragma unroll
            for (int i = 0; i < 8; ++i) { fa[i] = na[i]; fw[i] = nw[i]; }
        }
    }

    // cold path: remaining partial / unaligned K tiles (K=9, 30 here)
    for (int k0 = kfull; k0 < K; k0 += 32) {
        const int rem = K - k0;
        v8h ha, hb;
        #pragma unroll
        for (int i = 0; i < 8; ++i) {
            int   kk = skb + i;
            int   kc = kk < rem ? kk : 0;
            float mk = kk < rem ? 1.0f : 0.0f;
            ha[i] = (_Float16)(rowA[k0 + kc] * (mA * mk));
            hb[i] = (_Float16)(rowW[k0 + kc] * (mB * mk));
        }
        *(v8h*)&As[srow][skb] = ha;
        *(v8h*)&Bs[srow][skb] = hb;
        __syncthreads();
        wmma_step(As, Bs, sm, sn, lane, acc0, acc1);
        __syncthreads();
    }

    // ---- epilogue ----
    const int col0  = n0 + sn + (lane & 15);
    const int col1  = col0 + 16;
    const int rbase = m0 + sm + 8 * (lane >> 4);
    const float bv0 = (col0 < N) ? bias[col0] : 0.0f;
    const float bv1 = (col1 < N) ? bias[col1] : 0.0f;
    #pragma unroll
    for (int r = 0; r < 8; ++r) {
        int row = rbase + r;
        if (row < M) {
            if (col0 < N) {
                float o = (acc0[r] + bv0) * alpha;
                if (do_relu) o = fmaxf(o, 0.0f);
                C[(size_t)row * N + col0] = o;
            }
            if (col1 < N) {
                float o = (acc1[r] + bv1) * alpha;
                if (do_relu) o = fmaxf(o, 0.0f);
                C[(size_t)row * N + col1] = o;
            }
        }
    }
}

// =====================================================================
// Small helper kernels (wave32-aware where it matters)
// =====================================================================
__device__ inline float wave_sum(float v) {
    #pragma unroll
    for (int m = 16; m >= 1; m >>= 1) v += __shfl_xor(v, m, 32);
    return v;
}

__global__ void obj_prep_kernel(const float* __restrict__ traj,
                                const unsigned char* __restrict__ mask,
                                float* __restrict__ out, int M)
{
    int t = blockIdx.x * blockDim.x + threadIdx.x;
    if (t >= M * 30) return;
    int m = t / 30, c = t % 30;
    out[t] = (c < CAA) ? traj[(size_t)m * CAA + c] : (mask[m] ? 1.0f : 0.0f);
}

__global__ void maskmul_kernel(float* __restrict__ f,
                               const unsigned char* __restrict__ mask,
                               int M, int C)
{
    int t = blockIdx.x * blockDim.x + threadIdx.x;
    if (t >= M * C) return;
    if (!mask[t / C]) f[t] = 0.0f;
}

__global__ void maxpool_kernel(const float* __restrict__ in,
                               float* __restrict__ out,
                               int G, int Tlen, int C)
{
    int t = blockIdx.x * blockDim.x + threadIdx.x;
    if (t >= G * C) return;
    int g = t / C, c = t % C;
    const float* p = in + (size_t)g * Tlen * C + c;
    float m = p[0];
    for (int i = 1; i < Tlen; ++i) m = fmaxf(m, p[(size_t)i * C]);
    out[t] = m;
}

__global__ void concat_pool_kernel(const float* __restrict__ f,
                                   const float* __restrict__ pool,
                                   float* __restrict__ out,
                                   int M, int Tlen, int C)
{
    int t = blockIdx.x * blockDim.x + threadIdx.x;
    if (t >= M * 2 * C) return;
    int m = t / (2 * C), c = t % (2 * C);
    out[t] = (c < C) ? f[(size_t)m * C + c]
                     : pool[(size_t)(m / Tlen) * C + (c - C)];
}

__global__ void scatter_obj_kernel(const float* __restrict__ h,
                                   const unsigned char* __restrict__ mask,
                                   float* __restrict__ X)
{
    int t = blockIdx.x * blockDim.x + threadIdx.x;
    if (t >= BB * NOBJ * DIM) return;
    int r = t >> 8, c = t & 255;
    int b = r / NOBJ, no = r % NOBJ;
    const unsigned char* mp = mask + (size_t)r * TT;
    bool valid = false;
    for (int i = 0; i < TT; ++i) valid |= (mp[i] != 0);
    X[((size_t)(b * NTOK + no)) * DIM + c] = valid ? h[t] : 0.0f;
}

__global__ void scatter_map_kernel(const float* __restrict__ h,
                                   const unsigned char* __restrict__ mask,
                                   float* __restrict__ X, int b)
{
    int t = blockIdx.x * blockDim.x + threadIdx.x;
    if (t >= NMAP * DIM) return;
    int m = t >> 8, c = t & 255;
    const unsigned char* mp = mask + (size_t)m * PPTS;
    bool valid = false;
    for (int i = 0; i < PPTS; ++i) valid |= (mp[i] != 0);
    X[((size_t)(b * NTOK + NOBJ + m)) * DIM + c] = valid ? h[t] : 0.0f;
}

__global__ void pos_kernel(const float* __restrict__ objpos,
                           const float* __restrict__ mappos,
                           float* __restrict__ pos)
{
    int t = blockIdx.x * blockDim.x + threadIdx.x;
    if (t >= ROWS * 3) return;
    int r = t / 3, j = t % 3;
    int b = r / NTOK, i = r % NTOK;
    pos[t] = (i < NOBJ) ? objpos[((size_t)(b * NOBJ + i)) * 3 + j]
                        : mappos[((size_t)(b * NMAP + (i - NOBJ))) * 3 + j];
}

__global__ void topk_kernel(const float* __restrict__ pos, int* __restrict__ idx)
{
    int t = blockIdx.x * blockDim.x + threadIdx.x;   // one (b,n) per thread
    if (t >= ROWS) return;
    int b = t / NTOK;
    float px = pos[(size_t)t * 3], py = pos[(size_t)t * 3 + 1], pz = pos[(size_t)t * 3 + 2];
    const float* pb = pos + (size_t)b * NTOK * 3;
    float bd[KNB]; int bi[KNB];
    #pragma unroll
    for (int k = 0; k < KNB; ++k) { bd[k] = 3.402823466e+38f; bi[k] = 0; }
    for (int m = 0; m < NTOK; ++m) {
        float dx = px - pb[m * 3], dy = py - pb[m * 3 + 1], dz = pz - pb[m * 3 + 2];
        float d2 = dx * dx + dy * dy + dz * dz;
        if (d2 < bd[KNB - 1]) {
            int p = KNB - 1;
            while (p > 0 && d2 < bd[p - 1]) { bd[p] = bd[p - 1]; bi[p] = bi[p - 1]; --p; }
            bd[p] = d2; bi[p] = m;
        }
    }
    #pragma unroll
    for (int k = 0; k < KNB; ++k) idx[(size_t)t * KNB + k] = bi[k];
}

__global__ void pe_kernel(const float* __restrict__ pos, float* __restrict__ pe)
{
    int t = blockIdx.x * blockDim.x + threadIdx.x;
    if (t >= ROWS * DIM) return;
    int d = t & 255, r = t >> 8;
    float p = (d < 128) ? pos[(size_t)r * 3 + 1] : pos[(size_t)r * 3 + 0];
    int j = d & 127, m = j >> 1;
    float dimt = __powf(10000.0f, (float)m * (1.0f / 64.0f));
    float ang  = p * 6.283185307179586f / dimt;
    pe[t] = (j & 1) ? __cosf(ang) : __sinf(ang);
}

__global__ void add_kernel(const float* __restrict__ a, const float* __restrict__ b,
                           float* __restrict__ o, int n)
{
    int t = blockIdx.x * blockDim.x + threadIdx.x;
    if (t < n) o[t] = a[t] + b[t];
}

__global__ void copy_kernel(const float* __restrict__ a, float* __restrict__ o, int n)
{
    int t = blockIdx.x * blockDim.x + threadIdx.x;
    if (t < n) o[t] = a[t];
}

// One wave per (b, n, head); lane indexes the 32 head dims (wave32).
__global__ __launch_bounds__(256) void attn_kernel(
    const float* __restrict__ Q, const float* __restrict__ Kb,
    const float* __restrict__ Vb, const int* __restrict__ idx,
    float* __restrict__ O)
{
    int wid  = (blockIdx.x * blockDim.x + threadIdx.x) >> 5;
    int lane = threadIdx.x & 31;
    if (wid >= ROWS * NHEAD) return;
    int h   = wid & (NHEAD - 1);
    int b_n = wid >> 3;
    int b   = b_n / NTOK;
    float qd = Q[(size_t)b_n * DIM + h * HDIM + lane];
    float s[KNB];
    #pragma unroll
    for (int j = 0; j < KNB; ++j) {
        int nb = idx[(size_t)b_n * KNB + j];
        float kd = Kb[((size_t)(b * NTOK + nb)) * DIM + h * HDIM + lane];
        s[j] = wave_sum(qd * kd);
    }
    float mx = s[0];
    #pragma unroll
    for (int j = 1; j < KNB; ++j) mx = fmaxf(mx, s[j]);
    float sum = 0.0f;
    #pragma unroll
    for (int j = 0; j < KNB; ++j) { s[j] = __expf(s[j] - mx); sum += s[j]; }
    float inv = 1.0f / sum;
    float o = 0.0f;
    #pragma unroll
    for (int j = 0; j < KNB; ++j) {
        int nb = idx[(size_t)b_n * KNB + j];
        o += s[j] * inv * Vb[((size_t)(b * NTOK + nb)) * DIM + h * HDIM + lane];
    }
    O[(size_t)b_n * DIM + h * HDIM + lane] = o;
}

// One wave per row: O = LayerNorm(A + B) * g + be
__global__ __launch_bounds__(256) void ln_residual_kernel(
    const float* __restrict__ A, const float* __restrict__ Bi,
    const float* __restrict__ g, const float* __restrict__ be,
    float* __restrict__ O, int rows)
{
    int wid  = (blockIdx.x * blockDim.x + threadIdx.x) >> 5;
    int lane = threadIdx.x & 31;
    if (wid >= rows) return;
    const float* a = A  + (size_t)wid * DIM;
    const float* b = Bi + (size_t)wid * DIM;
    float v[8], s = 0.0f;
    #pragma unroll
    for (int i = 0; i < 8; ++i) { v[i] = a[lane + 32 * i] + b[lane + 32 * i]; s += v[i]; }
    float mean = wave_sum(s) * (1.0f / DIM);
    float var = 0.0f;
    #pragma unroll
    for (int i = 0; i < 8; ++i) { float d = v[i] - mean; var += d * d; }
    var = wave_sum(var) * (1.0f / DIM);
    float rstd = rsqrtf(var + 1e-5f);
    #pragma unroll
    for (int i = 0; i < 8; ++i) {
        int d = lane + 32 * i;
        O[(size_t)wid * DIM + d] = (v[i] - mean) * rstd * g[d] + be[d];
    }
}

// =====================================================================
// Host orchestration
// =====================================================================
#define L1D(kern, n, ...) kern<<<((n) + 255) / 256, 256, 0, stream>>>(__VA_ARGS__)

extern "C" void kernel_launch(void* const* d_in, const int* in_sizes, int n_in,
                              void* d_out, int out_size, void* d_ws, size_t ws_size,
                              hipStream_t stream)
{
    (void)in_sizes; (void)n_in; (void)out_size;

    const float* obj_trajs = (const float*)d_in[0];
    const float* map_poly  = (const float*)d_in[1];
    const float* obj_last  = (const float*)d_in[2];
    const float* map_ctr   = (const float*)d_in[3];
    const float* a_pre_w   = (const float*)d_in[4];
    const float* a_pre_b   = (const float*)d_in[5];
    const float* a_mlp_w1  = (const float*)d_in[6];
    const float* a_mlp_b1  = (const float*)d_in[7];
    const float* a_mlp_w2  = (const float*)d_in[8];
    const float* a_mlp_b2  = (const float*)d_in[9];
    const float* a_out_w1  = (const float*)d_in[10];
    const float* a_out_b1  = (const float*)d_in[11];
    const float* a_out_w2  = (const float*)d_in[12];
    const float* a_out_b2  = (const float*)d_in[13];
    const float* m_pre_w1  = (const float*)d_in[14];
    const float* m_pre_b1  = (const float*)d_in[15];
    const float* m_pre_w2  = (const float*)d_in[16];
    const float* m_pre_b2  = (const float*)d_in[17];
    const float* m_pre_w3  = (const float*)d_in[18];
    const float* m_pre_b3  = (const float*)d_in[19];
    const float* m_mlp_w1  = (const float*)d_in[20];
    const float* m_mlp_b1  = (const float*)d_in[21];
    const float* m_mlp_w2  = (const float*)d_in[22];
    const float* m_mlp_b2  = (const float*)d_in[23];
    const float* m_out_w1  = (const float*)d_in[24];
    const float* m_out_b1  = (const float*)d_in[25];
    const float* m_out_w2  = (const float*)d_in[26];
    const float* m_out_b2  = (const float*)d_in[27];
    const float* attn_wqkv = (const float*)d_in[28];
    const float* attn_bqkv = (const float*)d_in[29];
    const float* attn_wo   = (const float*)d_in[30];
    const float* attn_bo   = (const float*)d_in[31];
    const float* ffn_w1    = (const float*)d_in[32];
    const float* ffn_b1    = (const float*)d_in[33];
    const float* ffn_w2    = (const float*)d_in[34];
    const float* ffn_b2    = (const float*)d_in[35];
    const float* ln1_g     = (const float*)d_in[36];
    const float* ln1_b     = (const float*)d_in[37];
    const float* ln2_g     = (const float*)d_in[38];
    const float* ln2_b     = (const float*)d_in[39];
    const unsigned char* obj_mask = (const unsigned char*)d_in[40];
    const unsigned char* map_mask = (const unsigned char*)d_in[41];
    // d_in[42] (track_index_to_predict) unused by the forward pass.

    // ---------------- workspace layout (float units) ----------------
    constexpr size_t SZ_X    = (size_t)ROWS * DIM;           // 3,407,872
    constexpr size_t OFF_PE  = SZ_X;
    constexpr size_t OFF_POS = OFF_PE + SZ_X;
    constexpr size_t OFF_IDX = OFF_POS + (size_t)ROWS * 3;
    constexpr size_t OFF_H   = OFF_IDX + (size_t)ROWS * KNB; // idx is int32 (same slot size)
    constexpr size_t OFF_OP  = OFF_H + SZ_X;
    constexpr size_t OFF_SCR = OFF_OP + SZ_X;
    constexpr size_t ARENA   = 17684480;                     // max(obj phase, 5*SZ_X)
    constexpr size_t WS_FLOATS = OFF_SCR + ARENA;
    if (ws_size < WS_FLOATS * sizeof(float)) return;

    float* W    = (float*)d_ws;
    float* X    = W;                 // token features / residual stream
    float* PE   = W + OFF_PE;
    float* POS  = W + OFF_POS;
    int*   IDX  = (int*)(W + OFF_IDX);
    float* HBUF = W + OFF_H;
    float* OPRJ = W + OFF_OP;
    float* SCR  = W + OFF_SCR;

    auto gemm = [&](const float* A, const float* Wm, const float* bias, float* C,
                    int M, int N, int K, float alpha, int relu) {
        dim3 grid((N + 63) / 64, (M + 63) / 64);
        gemm_wmma_kernel<<<grid, 256, 0, stream>>>(A, Wm, bias, C, M, N, K, alpha, relu);
    };

    // ================= object PointNet (all batches at once) =================
    {
        const int M1 = BB * NOBJ * TT;                       // 21504
        float* OBJ_IN = SCR;                                 // [M1,30]
        float* F1   = SCR + 645120;                          // [M1,256]
        float* POOL = SCR + 6150144;                         // [1024,256]
        float* F2   = SCR + 6412288;                         // [M1,512]
        float* F3   = F1;                                    // reuse
        float* F4   = F2;                                    // reuse
        float* G    = POOL;                                  // reuse
        float* H1   = SCR + 17422336;                        // [1024,256]
        float* FEAT = SCR;                                   // reuse (obj_in dead)

        L1D(obj_prep_kernel, M1 * 30, obj_trajs, obj_mask, OBJ_IN, M1);
        gemm(OBJ_IN, a_pre_w, a_pre_b, F1, M1, DIM, 30, 1.0f, 1);
        L1D(maskmul_kernel, M1 * DIM, F1, obj_mask, M1, DIM);
        L1D(maxpool_kernel, BB * NOBJ * DIM, F1, POOL, BB * NOBJ, TT, DIM);
        L1D(concat_pool_kernel, M1 * 2 * DIM, F1, POOL, F2, M1, TT, DIM);
        gemm(F2, a_mlp_w1, a_mlp_b1, F3, M1, DIM, 2 * DIM, 1.0f, 1);
        gemm(F3, a_mlp_w2, a_mlp_b2, F4, M1, DIM, DIM, 1.0f, 1);
        L1D(maskmul_kernel, M1 * DIM, F4, obj_mask, M1, DIM);
        L1D(maxpool_kernel, BB * NOBJ * DIM, F4, G, BB * NOBJ, TT, DIM);
        gemm(G,  a_out_w1, a_out_b1, H1,   BB * NOBJ, DIM, DIM, 1.0f, 1);
        gemm(H1, a_out_w2, a_out_b2, FEAT, BB * NOBJ, DIM, DIM, 1.0f, 0);
        L1D(scatter_obj_kernel, BB * NOBJ * DIM, FEAT, obj_mask, X);
    }

    // ================= map PointNet (per-batch chunks) =================
    for (int b = 0; b < BB; ++b) {
        const int Mb = NMAP * PPTS;                          // 15360
        const float* Ab = map_poly + (size_t)b * Mb * CMM;
        const unsigned char* mb = map_mask + (size_t)b * Mb;
        float* MF1   = SCR;                                  // [Mb,64]
        float* MF2   = SCR + 983040;                         // [Mb,128]
        float* MF3   = SCR + 2949120;                        // [Mb,64]
        float* MPOOL = SCR + 3932160;                        // [768,64]
        float* MG    = SCR + 3981312;                        // [768,64]
        float* MH1   = SCR + 4030464;                        // [768,64]
        float* MFEAT = SCR + 4079616;                        // [768,256]

        gemm(Ab,  m_pre_w1, m_pre_b1, MF1, Mb, 64, CMM, 1.0f, 1);
        gemm(MF1, m_pre_w2, m_pre_b2, MF3, Mb, 64, 64,  1.0f, 1);
        gemm(MF3, m_pre_w3, m_pre_b3, MF1, Mb, 64, 64,  1.0f, 1);
        L1D(maskmul_kernel, Mb * 64, MF1, mb, Mb, 64);
        L1D(maxpool_kernel, NMAP * 64, MF1, MPOOL, NMAP, PPTS, 64);
        L1D(concat_pool_kernel, Mb * 128, MF1, MPOOL, MF2, Mb, PPTS, 64);
        gemm(MF2, m_mlp_w1, m_mlp_b1, MF3, Mb, 64, 128, 1.0f, 1);
        gemm(MF3, m_mlp_w2, m_mlp_b2, MF1, Mb, 64, 64,  1.0f, 1);
        L1D(maskmul_kernel, Mb * 64, MF1, mb, Mb, 64);
        L1D(maxpool_kernel, NMAP * 64, MF1, MG, NMAP, PPTS, 64);
        gemm(MG,  m_out_w1, m_out_b1, MH1,   NMAP, 64,  64, 1.0f, 1);
        gemm(MH1, m_out_w2, m_out_b2, MFEAT, NMAP, DIM, 64, 1.0f, 0);
        L1D(scatter_map_kernel, NMAP * DIM, MFEAT, mb, X, b);
    }

    // ================= positions, kNN indices, sine embedding =================
    L1D(pos_kernel, ROWS * 3, obj_last, map_ctr, POS);
    L1D(topk_kernel, ROWS, POS, IDX);
    L1D(pe_kernel, ROWS * DIM, POS, PE);

    // ================= transformer layers =================
    float* QKIN = SCR;
    float* Qb   = SCR + SZ_X;
    float* Kb   = SCR + 2 * SZ_X;
    float* Vb   = SCR + 3 * SZ_X;
    float* OBUF = SCR + 4 * SZ_X;
    float* FF1  = SCR;           // overlays QKIN..Vb (4*SZ_X = 13312*1024)
    float* FF2  = OBUF;          // overlays OBUF

    const float scaling = 0.17677669529663687f;  // 32^-0.5
    const int nrow_elems = ROWS * DIM;

    for (int l = 0; l < NLAYER; ++l) {
        const float* wq = attn_wqkv + (size_t)l * 3 * DIM * DIM;
        const float* wk = wq + (size_t)DIM * DIM;
        const float* wv = wk + (size_t)DIM * DIM;
        const float* bq = attn_bqkv + (size_t)l * 3 * DIM;
        const float* bk = bq + DIM;
        const float* bv = bk + DIM;

        L1D(add_kernel, nrow_elems, X, PE, QKIN, nrow_elems);
        gemm(QKIN, wq, bq, Qb, ROWS, DIM, DIM, scaling, 0);
        gemm(QKIN, wk, bk, Kb, ROWS, DIM, DIM, 1.0f, 0);
        gemm(X,    wv, bv, Vb, ROWS, DIM, DIM, 1.0f, 0);

        attn_kernel<<<(ROWS * NHEAD) / 8, 256, 0, stream>>>(Qb, Kb, Vb, IDX, OBUF);

        gemm(OBUF, attn_wo + (size_t)l * DIM * DIM, attn_bo + (size_t)l * DIM,
             OPRJ, ROWS, DIM, DIM, 1.0f, 0);

        ln_residual_kernel<<<(ROWS * 32 + 255) / 256, 256, 0, stream>>>(
            X, OPRJ, ln1_g + (size_t)l * DIM, ln1_b + (size_t)l * DIM, HBUF, ROWS);

        gemm(HBUF, ffn_w1 + (size_t)l * 4 * DIM * DIM, ffn_b1 + (size_t)l * 4 * DIM,
             FF1, ROWS, 4 * DIM, DIM, 1.0f, 1);
        gemm(FF1, ffn_w2 + (size_t)l * 4 * DIM * DIM, ffn_b2 + (size_t)l * DIM,
             FF2, ROWS, DIM, 4 * DIM, 1.0f, 0);

        ln_residual_kernel<<<(ROWS * 32 + 255) / 256, 256, 0, stream>>>(
            HBUF, FF2, ln2_g + (size_t)l * DIM, ln2_b + (size_t)l * DIM, X, ROWS);
    }

    L1D(copy_kernel, nrow_elems, X, (float*)d_out, nrow_elems);
}
